// GCNWithEdge_8899172237731
// MI455X (gfx1250) — matrix operand order repaired and checked by reading, not compiled, not run on hardware
//
#include <hip/hip_runtime.h>
#include <hip/hip_bf16.h>

typedef __attribute__((ext_vector_type(16))) _Float16 v16h;
typedef __attribute__((ext_vector_type(8)))  _Float16 v8h;
typedef __attribute__((ext_vector_type(8)))  float    v8f;

#define NNODES 50000
#define NEDGES 200000
#define MID    512         // fin*fout for both layers
#define MTILE  64          // edges per workgroup tile
#define H1_STRIDE 520      // padded LDS row stride in halves (16B aligned, bank-skewed)

// ---------------- small utility kernels ----------------

__global__ void zero_f32(float* __restrict__ p, int count) {
    int i = blockIdx.x * 256 + threadIdx.x;
    if (i < count) p[i] = 0.0f;
}

// W2 [512 x 512] f32 row-major (k,n) -> W2T f16 [n][k]
__global__ void cvt_w2(const float* __restrict__ w2, _Float16* __restrict__ w2t) {
    int idx = blockIdx.x * 256 + threadIdx.x;   // 262144 total
    int k = idx >> 9, n = idx & 511;
    w2t[(n << 9) | k] = (_Float16)w2[idx];
}

__global__ void deg_count(const int* __restrict__ dst, float* __restrict__ deg, int e) {
    int i = blockIdx.x * 256 + threadIdx.x;
    if (i < e) unsafeAtomicAdd(&deg[dst[i]], 1.0f);
}

// ---------------- fused edge kernel ----------------
// Per 64-edge tile:
//   phase1: h1 = lrelu(edge_attr @ W1 + b1)  -> LDS f16 [64][512]
//   phase2: h  = h1 @ W2 (+b2)               -> WMMA f32 accum
//            waves are COLUMN-split: each wave owns 8 col-blocks, loops the
//            4 row-blocks with 4 accumulators so each B fragment is loaded
//            from global exactly once per WG (512KB W2T read once per tile).
//            Col-blocks sharing the same output o are register-pre-reduced.
//   phase3: msg[e][o] += x[src[e]][i] * h[e][i*FOUT+o]   (ds_add_f32)
//   phase4: agg[dst[e]][o] += msg[e][o]                  (global_atomic_add_f32)
template<int FIN, int FOUT>
__global__ void edge_fused(const float* __restrict__ x,
                           const float* __restrict__ edge_attr,
                           const int*   __restrict__ src,
                           const int*   __restrict__ dstp,
                           const float* __restrict__ W1,   // [8][512]
                           const float* __restrict__ b1,   // [512]
                           const _Float16* __restrict__ W2T, // [512][512] (n-major)
                           const float* __restrict__ b2,   // [512]
                           float* __restrict__ agg)        // [N][FOUT]
{
    extern __shared__ char smem[];
    _Float16* sh_h1  = (_Float16*)smem;                               // 64*520*2 = 66560
    float*    sh_ea  = (float*)(smem + MTILE * H1_STRIDE * 2);        // 64*8*4   =  2048
    float*    sh_xs  = (float*)((char*)sh_ea + MTILE * 8 * 4);        // 64*FIN*4
    float*    sh_msg = (float*)((char*)sh_xs + MTILE * FIN * 4);      // 64*FOUT*4
    int*      sh_dst = (int*)((char*)sh_msg + MTILE * FOUT * 4);      // 64*4

    const int t = threadIdx.x;                 // 0..127  (4 waves)
    const long long e0 = (long long)blockIdx.x * MTILE;

    // ---- stage edge_attr, src-features, dst; zero msg ----
    for (int i = t; i < MTILE * 8; i += 128) sh_ea[i] = edge_attr[e0 * 8 + i];
    if (t < MTILE) {
        int s = src[e0 + t];
        sh_dst[t] = dstp[e0 + t];
        #pragma unroll
        for (int i = 0; i < FIN; ++i) sh_xs[t * FIN + i] = x[(long long)s * FIN + i];
    }
    for (int i = t; i < MTILE * FOUT; i += 128) sh_msg[i] = 0.0f;
    __syncthreads();

    // ---- phase 1: h1 (K=8 GEMM on VALU), 4 columns per thread ----
    {
        float w[4][8], bb[4];
        #pragma unroll
        for (int j = 0; j < 4; ++j) {
            int c = t + 128 * j;
            bb[j] = b1[c];
            #pragma unroll
            for (int d = 0; d < 8; ++d) w[j][d] = W1[d * 512 + c];
        }
        for (int r = 0; r < MTILE; ++r) {
            float ea[8];
            #pragma unroll
            for (int d = 0; d < 8; ++d) ea[d] = sh_ea[r * 8 + d];
            #pragma unroll
            for (int j = 0; j < 4; ++j) {
                float acc = bb[j];
                #pragma unroll
                for (int d = 0; d < 8; ++d) acc += ea[d] * w[j][d];
                acc = acc >= 0.0f ? acc : 0.01f * acc;           // lrelu
                sh_h1[r * H1_STRIDE + t + 128 * j] = (_Float16)acc;
            }
        }
    }
    __syncthreads();

    // ---- phase 2+3: WMMA over K=512, fold into msg ----
    const int wave  = t >> 5;
    const int lane  = t & 31;
    const int lhalf = (lane >= 16) ? 1 : 0;
    const int l15   = lane & 15;

    constexpr int NGROUP = FOUT / 16;          // col-blocks with distinct o per wave
    constexpr int JPER   = 8 / NGROUP;         // col-blocks sharing one o

    for (int g = 0; g < NGROUP; ++g) {
        float macc[4][8];
        #pragma unroll
        for (int rb = 0; rb < 4; ++rb)
            #pragma unroll
            for (int r = 0; r < 8; ++r) macc[rb][r] = 0.0f;

        for (int jj = 0; jj < JPER; ++jj) {
            const int jn = g + jj * NGROUP;     // nb % NGROUP == g (wave*8 is even)
            const int nb = wave * 8 + jn;       // 0..31
            const int n  = nb * 16 + l15;       // this lane's output column
            const _Float16* bcol = &W2T[(long long)n << 9];
            // prefetch next column strip (global_prefetch_b8)
            __builtin_prefetch(bcol + 512 * 16 * NGROUP, 0, 3);

            v8f acc[4];
            #pragma unroll
            for (int q = 0; q < 4; ++q) acc[q] = v8f{0.f,0.f,0.f,0.f,0.f,0.f,0.f,0.f};

            for (int ks = 0; ks < 16; ++ks) {   // K steps of 32
                const int kb = ks * 32;
                // B fragment: lanes 0-15: K kb..kb+15, lanes 16-31: +16
                const int kB = kb + lhalf * 16;
                v8h blo = *(const v8h*)&bcol[kB];
                v8h bhi = *(const v8h*)&bcol[kB + 8];
                v16h b;
                #pragma unroll
                for (int j = 0; j < 8; ++j) { b[j] = blo[j]; b[j + 8] = bhi[j]; }

                // batch-load all 4 A fragments, then issue 4 wmma
                const int ka = kb + lhalf * 16;
                v16h a[4];
                #pragma unroll
                for (int rb = 0; rb < 4; ++rb) {
                    const int arow = rb * 16 + l15;
                    v8h alo = *(const v8h*)&sh_h1[arow * H1_STRIDE + ka];
                    v8h ahi = *(const v8h*)&sh_h1[arow * H1_STRIDE + ka + 8];
                    #pragma unroll
                    for (int j = 0; j < 8; ++j) { a[rb][j] = alo[j]; a[rb][j + 8] = ahi[j]; }
                }
                #pragma unroll
                for (int rb = 0; rb < 4; ++rb)
                    acc[rb] = __builtin_amdgcn_wmma_f32_16x16x32_f16(
                        false, a[rb], false, b, (short)0, acc[rb], false, false);
            }

            // register pre-reduction over i (same o across jj group)
            const float b2n = b2[n];
            const int ii = n / FOUT;
            #pragma unroll
            for (int rb = 0; rb < 4; ++rb) {
                #pragma unroll
                for (int r = 0; r < 8; ++r) {
                    const int row = rb * 16 + r + lhalf * 8;
                    macc[rb][r] += (acc[rb][r] + b2n) * sh_xs[row * FIN + ii];
                }
            }
        }

        // phase 3: single LDS atomic per (row, o) group
        const int o = ((wave * 8 + g) * 16 + l15) % FOUT;
        #pragma unroll
        for (int rb = 0; rb < 4; ++rb) {
            #pragma unroll
            for (int r = 0; r < 8; ++r) {
                const int row = rb * 16 + r + lhalf * 8;
                atomicAdd(&sh_msg[row * FOUT + o], macc[rb][r]);   // ds_add_f32
            }
        }
    }
    __syncthreads();

    // ---- phase 4: scatter-add messages to agg[dst] ----
    for (int idx = t; idx < MTILE * FOUT; idx += 128) {
        const int r = idx / FOUT, o = idx % FOUT;
        unsafeAtomicAdd(&agg[(long long)sh_dst[r] * FOUT + o], sh_msg[idx]);
    }
}

// ---------------- node update: agg/deg + x@root + bias, accumulate BN stats ----------------
template<int FIN, int FOUT>
__global__ void node_update(const float* __restrict__ agg, const float* __restrict__ deg,
                            const float* __restrict__ x,   const float* __restrict__ root,
                            const float* __restrict__ bias, float* __restrict__ pre,
                            float* __restrict__ stats, int nnodes)
{
    const int t   = threadIdx.x;
    const int gid = blockIdx.x * 256 + t;
    const int n   = gid / FOUT, o = gid % FOUT;
    float val = 0.0f;
    if (n < nnodes) {
        float d = deg[n]; d = d > 1.0f ? d : 1.0f;
        float acc = agg[(long long)n * FOUT + o] / d + bias[o];
        #pragma unroll
        for (int i = 0; i < FIN; ++i) acc += x[(long long)n * FIN + i] * root[i * FOUT + o];
        pre[(long long)n * FOUT + o] = acc;
        val = acc;
    }
    __shared__ float s1[256], s2[256];
    s1[t] = val; s2[t] = val * val;
    __syncthreads();
    for (int off = 128; off >= FOUT; off >>= 1) {
        if (t < off) { s1[t] += s1[t + off]; s2[t] += s2[t + off]; }
        __syncthreads();
    }
    if (t < FOUT) {
        unsafeAtomicAdd(&stats[t],        s1[t]);
        unsafeAtomicAdd(&stats[FOUT + t], s2[t]);
    }
}

template<int FOUT, bool LRELU>
__global__ void bn_apply(const float* __restrict__ pre, const float* __restrict__ stats,
                         const float* __restrict__ gamma, const float* __restrict__ beta,
                         float* __restrict__ out, int nnodes)
{
    const int gid = blockIdx.x * 256 + threadIdx.x;
    const int n = gid / FOUT, o = gid % FOUT;
    if (n >= nnodes) return;
    const float invN = 1.0f / (float)nnodes;
    float mu = stats[o] * invN;
    float e2 = stats[FOUT + o] * invN;
    float var = e2 - mu * mu;
    float rs = rsqrtf(var + 1e-5f);
    float y = (pre[(long long)n * FOUT + o] - mu) * rs * gamma[o] + beta[o];
    if (LRELU) y = y >= 0.0f ? y : 0.01f * y;
    out[(long long)n * FOUT + o] = y;
}

// ---------------- host ----------------
extern "C" void kernel_launch(void* const* d_in, const int* in_sizes, int n_in,
                              void* d_out, int out_size, void* d_ws, size_t ws_size,
                              hipStream_t stream) {
    (void)in_sizes; (void)n_in; (void)out_size; (void)ws_size;
    const float* x      = (const float*)d_in[0];
    const float* ea     = (const float*)d_in[1];
    const int*   eidx   = (const int*)d_in[2];
    const float* W1_0   = (const float*)d_in[3];
    const float* b1_0   = (const float*)d_in[4];
    const float* W2_0   = (const float*)d_in[5];
    const float* b2_0   = (const float*)d_in[6];
    const float* root_0 = (const float*)d_in[7];
    const float* bias_0 = (const float*)d_in[8];
    const float* gamma_0= (const float*)d_in[9];
    const float* beta_0 = (const float*)d_in[10];
    const float* W1_1   = (const float*)d_in[11];
    const float* b1_1   = (const float*)d_in[12];
    const float* W2_1   = (const float*)d_in[13];
    const float* b2_1   = (const float*)d_in[14];
    const float* root_1 = (const float*)d_in[15];
    const float* bias_1 = (const float*)d_in[16];
    const float* gamma_1= (const float*)d_in[17];
    const float* beta_1 = (const float*)d_in[18];

    const int* srcp = eidx;
    const int* dstp = eidx + NEDGES;

    char* base = (char*)d_ws;
    size_t off = 0;
    auto alloc = [&](size_t bytes) { size_t c = off; off = (off + bytes + 255) & ~(size_t)255; return c; };

    size_t o_w2t0 = alloc(512 * 512 * 2);
    size_t o_w2t1 = alloc(512 * 512 * 2);
    size_t zbase  = off;
    size_t o_deg   = alloc((size_t)NNODES * 4);
    size_t o_agg0  = alloc((size_t)NNODES * 32 * 4);
    size_t o_agg1  = alloc((size_t)NNODES * 16 * 4);
    size_t o_st0   = alloc(64 * 4);
    size_t o_st1   = alloc(32 * 4);
    size_t zcount  = (off - zbase) / 4;
    size_t o_pre0  = alloc((size_t)NNODES * 32 * 4);
    size_t o_hmid  = alloc((size_t)NNODES * 32 * 4);
    size_t o_pre1  = alloc((size_t)NNODES * 16 * 4);

    _Float16* w2t0 = (_Float16*)(base + o_w2t0);
    _Float16* w2t1 = (_Float16*)(base + o_w2t1);
    float* deg  = (float*)(base + o_deg);
    float* agg0 = (float*)(base + o_agg0);
    float* agg1 = (float*)(base + o_agg1);
    float* st0  = (float*)(base + o_st0);
    float* st1  = (float*)(base + o_st1);
    float* pre0 = (float*)(base + o_pre0);
    float* hmid = (float*)(base + o_hmid);
    float* pre1 = (float*)(base + o_pre1);

    // 1. zero accumulators
    zero_f32<<<(int)((zcount + 255) / 256), 256, 0, stream>>>((float*)(base + zbase), (int)zcount);
    // 2. weight convert/transpose to f16
    cvt_w2<<<1024, 256, 0, stream>>>(W2_0, w2t0);
    cvt_w2<<<1024, 256, 0, stream>>>(W2_1, w2t1);
    // 3. degree
    deg_count<<<(NEDGES + 255) / 256, 256, 0, stream>>>(dstp, deg, NEDGES);

    const size_t shbytes = MTILE * H1_STRIDE * 2   // h1 f16
                         + MTILE * 8 * 4           // edge_attr
                         + MTILE * 48 * 4          // xs + msg (FIN+FOUT = 48)
                         + MTILE * 4;              // dst
    const int ngrid0 = (NNODES * 32 + 255) / 256;
    const int ngrid1 = (NNODES * 16 + 255) / 256;

    // ---- layer 0: fin=16, fout=32 ----
    edge_fused<16, 32><<<NEDGES / MTILE, 128, shbytes, stream>>>(
        x, ea, srcp, dstp, W1_0, b1_0, w2t0, b2_0, agg0);
    node_update<16, 32><<<ngrid0, 256, 0, stream>>>(agg0, deg, x, root_0, bias_0, pre0, st0, NNODES);
    bn_apply<32, true><<<ngrid0, 256, 0, stream>>>(pre0, st0, gamma_0, beta_0, hmid, NNODES);

    // ---- layer 1: fin=32, fout=16 ----
    edge_fused<32, 16><<<NEDGES / MTILE, 128, shbytes, stream>>>(
        hmid, ea, srcp, dstp, W1_1, b1_1, w2t1, b2_1, agg1);
    node_update<32, 16><<<ngrid1, 256, 0, stream>>>(agg1, deg, hmid, root_1, bias_1, pre1, st1, NNODES);
    bn_apply<16, false><<<ngrid1, 256, 0, stream>>>(pre1, st1, gamma_1, beta_1, (float*)d_out, NNODES);
}
